// TSTEncoderLayer_15083925143885
// MI455X (gfx1250) — matrix operand order repaired and checked by reading, not compiled, not run on hardware
//
#include <hip/hip_runtime.h>
#include <hip/hip_bf16.h>
#include <math.h>

// ---------------------------------------------------------------------------
// Problem constants (from reference):  B=256 L=96 D=512 H=8 HD=64 DF=2048
// ---------------------------------------------------------------------------
#define BB    256
#define LL    96
#define DD    512
#define HH    8
#define HDIM  64
#define DFF   2048
#define MROWS (BB * LL)          // 24576 rows for all "token" GEMMs
#define SCALE 0.125f             // HD^-0.5 = 64^-0.5
#define EPSLN 1e-5f

typedef __attribute__((ext_vector_type(16))) _Float16 v16h;
typedef __attribute__((ext_vector_type(8)))  float    v8f;

// ---------------------------------------------------------------------------
// CDNA5 async global->LDS staging (ASYNCcnt path), with portable fallback.
// Probe result (round 2 stderr): the builtin's pointer params are
//   arg0: int vector_size(16) addrspace(1)*   (global, non-const)
//   arg1: int vector_size(16) addrspace(3)*   (LDS)
//   arg2/3: immediate ints (offset, cpol)
// ---------------------------------------------------------------------------
#if defined(__HIP_DEVICE_COMPILE__) && __has_builtin(__builtin_amdgcn_global_load_async_to_lds_b128)
#define ASYNC_LDS 1
#else
#define ASYNC_LDS 0
#endif

typedef int v4i_vec __attribute__((vector_size(16)));
#define TO_GLOBAL4(p) ((__attribute__((address_space(1))) v4i_vec*)(void*)(p))
#define TO_LDS4(p)    ((__attribute__((address_space(3))) v4i_vec*)(void*)(p))

template <int N>
__device__ __forceinline__ void wait_async() {
#if defined(__HIP_DEVICE_COMPILE__) && __has_builtin(__builtin_amdgcn_s_wait_asynccnt)
  __builtin_amdgcn_s_wait_asynccnt(N);
#elif defined(__HIP_DEVICE_COMPILE__)
  asm volatile("s_wait_asynccnt %0" ::"i"(N) : "memory");
#endif
}

// ---------------------------------------------------------------------------
// WMMA fragment helpers (CDNA5 16x16x32 f16 -> f32).
// A layout (ISA 7.12.2, 16-bit A 16x32): lane = {row=l&15, half=l>>4};
//   dwords 0..3 hold K = half*8 + {0..7}, dwords 4..7 hold K = 16+half*8+{0..7}
// B layout (same K order as A): lane = {col=l&15,
//   half=l>>4}; dwords 0..3 hold K = half*8 + {0..7}, dwords 4..7 hold K = 16+half*8+{0..7}
// Both loaders read K-contiguous storage (row-major A, B stored as [N][K]).
// ---------------------------------------------------------------------------
__device__ __forceinline__ v16h load_a_frag(const _Float16* base, int stride, int lane) {
  const int row = lane & 15, half = lane >> 4;
  const _Float16* p = base + (size_t)row * stride + half * 8;
  union { v16h v; unsigned u[8]; } f;
#pragma unroll
  for (int i = 0; i < 4; ++i) f.u[i]     = *(const unsigned*)(p + 2 * i);
#pragma unroll
  for (int i = 0; i < 4; ++i) f.u[4 + i] = *(const unsigned*)(p + 16 + 2 * i);
  return f.v;
}

__device__ __forceinline__ v16h load_b_frag(const _Float16* base, int stride, int lane) {
  const int col = lane & 15, half = lane >> 4;
  const _Float16* p = base + (size_t)col * stride + half * 8;
  union { v16h v; unsigned u[8]; } f;
#pragma unroll
  for (int i = 0; i < 8; ++i) f.u[i] = *(const unsigned*)(p + (i >> 2) * 16 + 2 * (i & 3));
  return f.v;
}

__device__ __forceinline__ v8f wmma16(v16h a, v16h b, v8f c) {
  return __builtin_amdgcn_wmma_f32_16x16x32_f16(
      /*neg_a=*/false, a, /*neg_b=*/false, b,
      /*c_mod=*/(short)0, c, /*reuse_a=*/false, /*reuse_b=*/false);
}

// ---------------------------------------------------------------------------
// f32 -> f16 conversion (straight)
// ---------------------------------------------------------------------------
__global__ __launch_bounds__(256) void cvt_f32_f16_kernel(
    const float* __restrict__ in, _Float16* __restrict__ out, int n) {
  int i = blockIdx.x * 256 + threadIdx.x;
  if (i < n) out[i] = (_Float16)in[i];
}

// ---------------------------------------------------------------------------
// f32 [K][N] -> f16 [N][K] transpose+convert, 32x32 LDS tiles.
// Makes every GEMM operand K-contiguous so LDS staging is pure b128 copies.
// ---------------------------------------------------------------------------
__global__ __launch_bounds__(256) void cvtT_f32_f16_kernel(
    const float* __restrict__ in, _Float16* __restrict__ out, int K, int N) {
  __shared__ float t[32][33];
  const int n0 = blockIdx.x * 32, k0 = blockIdx.y * 32;
  const int tx = threadIdx.x & 31, ty = threadIdx.x >> 5;  // 32 x 8
#pragma unroll
  for (int i = 0; i < 32; i += 8)
    t[ty + i][tx] = in[(size_t)(k0 + ty + i) * N + n0 + tx];
  __syncthreads();
#pragma unroll
  for (int i = 0; i < 32; i += 8)
    out[(size_t)(n0 + ty + i) * K + k0 + tx] = (_Float16)t[tx][ty + i];
}

// ---------------------------------------------------------------------------
// Tiled f16 GEMM:  C[M,N] = A[M,K] @ BT[N,K]^T + bias, M = 24576.
// Workgroup tile 128x128, 8 waves (4Mx2N), wave tile 32x64 = 2x4 WMMA accs.
// K staged in 32-wide slabs; double-buffered async global->LDS copies.
// EPI: 0 = f32 out (+bias)                       (attn proj, FFN2)
//      1 = f16 out scattered to [B,H,L,HD] (+bias)  (Q/K/V)
//      2 = f16 out with exact GELU (+bias)          (FFN1)
// ---------------------------------------------------------------------------
template <int EPI>
__global__ __launch_bounds__(256) void gemm_f16_kernel(
    const _Float16* __restrict__ A, const _Float16* __restrict__ BT,
    const float* __restrict__ bias, float* __restrict__ outF,
    _Float16* __restrict__ outH, int N, int K) {
  __shared__ alignas(16) _Float16 As[2][128][40];   // [buf][M][K32] (80B rows)
  __shared__ alignas(16) _Float16 Bs[2][128][40];   // [buf][N][K32]

  const int tid  = threadIdx.x;
  const int lane = tid & 31;
  const int w    = tid >> 5;
  const int wm   = (w & 3) * 32;        // wave M offset inside 128
  const int wn   = (w >> 2) * 64;       // wave N offset inside 128
  const int m0   = blockIdx.x * 128;
  const int n0   = blockIdx.y * 128;

  // stage one 128x32 slab of A and of BT into LDS buffer `buf`
  auto stage = [&](int kb, int buf) {
    const int k0 = kb << 5;
#pragma unroll
    for (int i = 0; i < 2; ++i) {
      const int c   = tid + (i << 8);     // 0..511
      const int row = c >> 2;
      const int cb  = (c & 3) << 3;       // 0,8,16,24 halves
      const _Float16* ga = A  + (size_t)(m0 + row) * K + k0 + cb;
      const _Float16* gb = BT + (size_t)(n0 + row) * K + k0 + cb;
#if ASYNC_LDS
      __builtin_amdgcn_global_load_async_to_lds_b128(
          TO_GLOBAL4(ga), TO_LDS4(&As[buf][row][cb]), 0, 0);
      __builtin_amdgcn_global_load_async_to_lds_b128(
          TO_GLOBAL4(gb), TO_LDS4(&Bs[buf][row][cb]), 0, 0);
#else
      *(uint4*)(&As[buf][row][cb]) = *(const uint4*)ga;
      *(uint4*)(&Bs[buf][row][cb]) = *(const uint4*)gb;
      __builtin_prefetch(ga + 32, 0, 1);
      __builtin_prefetch(gb + 32, 0, 1);
#endif
    }
  };

  const v8f vz = {};
  v8f acc[2][4];
#pragma unroll
  for (int i = 0; i < 2; ++i)
#pragma unroll
    for (int j = 0; j < 4; ++j) acc[i][j] = vz;

  const int nk = K >> 5;
  stage(0, 0);
  for (int kb = 0; kb < nk; ++kb) {
    const int buf = kb & 1;
    if (kb + 1 < nk) {
      stage(kb + 1, buf ^ 1);   // overlap next slab's DMA with this compute
      wait_async<4>();          // 4 outstanding = next slab; current complete
    } else {
      wait_async<0>();
    }
    __syncthreads();            // all threads' slab-kb writes visible

    v16h af[2], bf[4];
#pragma unroll
    for (int i = 0; i < 2; ++i) af[i] = load_a_frag(&As[buf][wm + i * 16][0], 40, lane);
#pragma unroll
    for (int j = 0; j < 4; ++j) bf[j] = load_b_frag(&Bs[buf][wn + j * 16][0], 40, lane);
#pragma unroll
    for (int i = 0; i < 2; ++i)
#pragma unroll
      for (int j = 0; j < 4; ++j) acc[i][j] = wmma16(af[i], bf[j], acc[i][j]);

    __syncthreads();            // reads done before this buf is re-targeted
  }

  // ---- epilogue: C layout lane={col=l&15, half=l>>4}, dword r -> row half*8+r
  const int half = lane >> 4, cl = lane & 15;
#pragma unroll
  for (int i = 0; i < 2; ++i)
#pragma unroll
    for (int j = 0; j < 4; ++j) {
      const int n  = n0 + wn + j * 16 + cl;
      const float bv = bias[n];
#pragma unroll
      for (int r = 0; r < 8; ++r) {
        const int m = m0 + wm + i * 16 + half * 8 + r;
        float val = acc[i][j][r] + bv;
        if (EPI == 0) {
          outF[(size_t)m * N + n] = val;
        } else if (EPI == 1) {
          const int b = m / LL, l = m - b * LL;
          const int hh = n >> 6, hd = n & 63;
          outH[(((size_t)b * HH + hh) * LL + l) * HDIM + hd] = (_Float16)val;
        } else {
          float g = 0.5f * val * (1.0f + erff(val * 0.70710678118654752f));
          outH[(size_t)m * N + n] = (_Float16)g;
        }
      }
    }
}

// ---------------------------------------------------------------------------
// Attention: one workgroup (6 waves, 192 threads) per (b,h).
// Wave w owns query rows [w*16, w*16+16).
//   S = q @ k^T  (k's natural [Lk,HD] row-major layout == packed-[N][K] B)
//   S *= matrix * SCALE  -> write `scores` output
//   row softmax in registers (shfl within 16-lane half-groups)
//   P -> LDS (f16), V^T -> LDS, O = P @ V via WMMA, scatter to [B,L,H*HD]
// ---------------------------------------------------------------------------
__global__ __launch_bounds__(192) void attn_kernel(
    const _Float16* __restrict__ q, const _Float16* __restrict__ kmat,
    const _Float16* __restrict__ v, const float* __restrict__ matrix,
    float* __restrict__ scores, _Float16* __restrict__ attn_out) {
  __shared__ _Float16 Pl[96][104];   // softmax probs, K-contiguous for A frags
  __shared__ _Float16 vT[64][104];   // V transposed: [d][lk], K-contiguous B

  const int bh = blockIdx.x;
  const int b  = bh >> 3;            // H == 8
  const int h  = bh & 7;
  const size_t base = (size_t)bh * LL * HDIM;

  const int tid = threadIdx.x, lane = tid & 31, w = tid >> 5;

  // ---- stage V^T: 96x64 halves = 768 chunks of 16B over 192 threads ----
#pragma unroll
  for (int i = 0; i < 4; ++i) {
    int c  = tid + i * 192;
    int lk = c >> 3;
    int cb = (c & 7) << 3;
    uint4 d = *(const uint4*)(v + base + (size_t)lk * HDIM + cb);
    const _Float16* hp = (const _Float16*)&d;
#pragma unroll
    for (int j = 0; j < 8; ++j) vT[cb + j][lk] = hp[j];
  }

  // ---- S = q @ k^T ----
  const int m0 = w * 16;
  const v8f vz = {};
  v8f s[6] = {vz, vz, vz, vz, vz, vz};
#pragma unroll
  for (int t = 0; t < 2; ++t) {
    const int k0 = t * 32;
    v16h a = load_a_frag(q + base + (size_t)m0 * HDIM + k0, HDIM, lane);
#pragma unroll
    for (int j = 0; j < 6; ++j) {
      v16h bb = load_b_frag(kmat + base + (size_t)(j * 16) * HDIM + k0, HDIM, lane);
      s[j] = wmma16(a, bb, s[j]);
    }
  }

  const int half = lane >> 4, cl = lane & 15;
  const size_t srow = (size_t)bh * LL;

  // ---- elementwise score matrix * SCALE, emit scores output ----
#pragma unroll
  for (int j = 0; j < 6; ++j) {
    const int n = j * 16 + cl;
#pragma unroll
    for (int r = 0; r < 8; ++r) {
      const int m = m0 + half * 8 + r;
      float val = s[j][r] * matrix[(srow + m) * LL + n] * SCALE;
      s[j][r] = val;
      scores[(srow + m) * LL + n] = val;
    }
  }

  // ---- softmax over the 96 key positions of each row ----
#pragma unroll
  for (int r = 0; r < 8; ++r) {
    float mx = -3.4e38f;
#pragma unroll
    for (int j = 0; j < 6; ++j) mx = fmaxf(mx, s[j][r]);
#pragma unroll
    for (int off = 1; off < 16; off <<= 1) mx = fmaxf(mx, __shfl_xor(mx, off, 32));
    float sum = 0.f;
#pragma unroll
    for (int j = 0; j < 6; ++j) {
      float e = __expf(s[j][r] - mx);
      s[j][r] = e;
      sum += e;
    }
#pragma unroll
    for (int off = 1; off < 16; off <<= 1) sum += __shfl_xor(sum, off, 32);
    const float inv = 1.0f / sum;
    const int m = m0 + half * 8 + r;
#pragma unroll
    for (int j = 0; j < 6; ++j) {
      s[j][r] *= inv;
      Pl[m][j * 16 + cl] = (_Float16)s[j][r];
    }
  }
  __syncthreads();

  // ---- O = P @ V : [96 x 64], K = 96 ----
  v8f o[4] = {vz, vz, vz, vz};
#pragma unroll
  for (int t = 0; t < 3; ++t) {
    const int k0 = t * 32;
    v16h a = load_a_frag(&Pl[m0][k0], 104, lane);
#pragma unroll
    for (int j = 0; j < 4; ++j) {
      v16h bb = load_b_frag(&vT[j * 16][k0], 104, lane);
      o[j] = wmma16(a, bb, o[j]);
    }
  }
#pragma unroll
  for (int j = 0; j < 4; ++j) {
    const int d = j * 16 + cl;
#pragma unroll
    for (int r = 0; r < 8; ++r) {
      const int m = m0 + half * 8 + r;
      attn_out[((size_t)(b * LL + m)) * DD + h * HDIM + d] = (_Float16)o[j][r];
    }
  }
}

// ---------------------------------------------------------------------------
// x = LayerNorm(a + b) * gamma + beta ; one block (256 thr) per 512-elem row.
// Writes f32 (residual chain) and optionally f16 (next GEMM operand).
// ---------------------------------------------------------------------------
__global__ __launch_bounds__(256) void add_ln_kernel(
    const float* __restrict__ a, const float* __restrict__ bres,
    const float* __restrict__ gamma, const float* __restrict__ beta,
    float* __restrict__ outF, _Float16* __restrict__ outH) {
  __shared__ float red[8];
  const int m = blockIdx.x, tid = threadIdx.x;
  const int lane = tid & 31, w = tid >> 5;
  const size_t rb = (size_t)m * DD;

  float t0 = a[rb + tid] + bres[rb + tid];
  float t1 = a[rb + tid + 256] + bres[rb + tid + 256];

  float s = t0 + t1;
#pragma unroll
  for (int off = 1; off < 32; off <<= 1) s += __shfl_xor(s, off, 32);
  if (lane == 0) red[w] = s;
  __syncthreads();
  float tot = 0.f;
#pragma unroll
  for (int i = 0; i < 8; ++i) tot += red[i];
  const float mean = tot * (1.0f / DD);
  __syncthreads();

  const float d0 = t0 - mean, d1 = t1 - mean;
  float vs = d0 * d0 + d1 * d1;
#pragma unroll
  for (int off = 1; off < 32; off <<= 1) vs += __shfl_xor(vs, off, 32);
  if (lane == 0) red[w] = vs;
  __syncthreads();
  float vtot = 0.f;
#pragma unroll
  for (int i = 0; i < 8; ++i) vtot += red[i];
  const float inv = rsqrtf(vtot * (1.0f / DD) + EPSLN);

  const float o0 = d0 * inv * gamma[tid] + beta[tid];
  const float o1 = d1 * inv * gamma[tid + 256] + beta[tid + 256];
  outF[rb + tid] = o0;
  outF[rb + tid + 256] = o1;
  if (outH != nullptr) {
    outH[rb + tid] = (_Float16)o0;
    outH[rb + tid + 256] = (_Float16)o1;
  }
}

// ---------------------------------------------------------------------------
// Host orchestration
// ---------------------------------------------------------------------------
extern "C" void kernel_launch(void* const* d_in, const int* in_sizes, int n_in,
                              void* d_out, int out_size, void* d_ws, size_t ws_size,
                              hipStream_t stream) {
  const float* src    = (const float*)d_in[0];
  const float* Wq     = (const float*)d_in[1];
  const float* bq     = (const float*)d_in[2];
  const float* Wk     = (const float*)d_in[3];
  const float* bk     = (const float*)d_in[4];
  const float* Wv     = (const float*)d_in[5];
  const float* bv     = (const float*)d_in[6];
  const float* matrix = (const float*)d_in[7];
  const float* Wo     = (const float*)d_in[8];
  const float* bo     = (const float*)d_in[9];
  const float* ln1_g  = (const float*)d_in[10];
  const float* ln1_b  = (const float*)d_in[11];
  const float* W1     = (const float*)d_in[12];
  const float* b1     = (const float*)d_in[13];
  const float* W2     = (const float*)d_in[14];
  const float* b2     = (const float*)d_in[15];
  const float* ln2_g  = (const float*)d_in[16];
  const float* ln2_b  = (const float*)d_in[17];

  float* y_out      = (float*)d_out;                       // [B,L,D]
  float* scores_out = y_out + (size_t)BB * LL * DD;        // [B,H,L,L]

  // ---- workspace carve-up ----
  char* ws = (char*)d_ws;
  size_t off = 0;
  auto alloc = [&](size_t bytes) -> char* {
    char* p = ws + off;
    off = (off + bytes + 255) & ~(size_t)255;
    return p;
  };
  _Float16* hWqT = (_Float16*)alloc((size_t)DD * DD * 2);    // [N][K] = W^T
  _Float16* hWkT = (_Float16*)alloc((size_t)DD * DD * 2);
  _Float16* hWvT = (_Float16*)alloc((size_t)DD * DD * 2);
  _Float16* hWoT = (_Float16*)alloc((size_t)DD * DD * 2);
  _Float16* hW1T = (_Float16*)alloc((size_t)DD * DFF * 2);   // [2048][512]
  _Float16* hW2T = (_Float16*)alloc((size_t)DFF * DD * 2);   // [512][2048]
  _Float16* hsrc = (_Float16*)alloc((size_t)MROWS * DD * 2);
  _Float16* qb   = (_Float16*)alloc((size_t)MROWS * DD * 2); // [B,H,L,HD]
  _Float16* kb_  = (_Float16*)alloc((size_t)MROWS * DD * 2);
  _Float16* vb   = (_Float16*)alloc((size_t)MROWS * DD * 2);
  _Float16* aob  = (_Float16*)alloc((size_t)MROWS * DD * 2); // attn out [B,L,D]
  float*    proj = (float*)   alloc((size_t)MROWS * DD * 4); // also x after LN1
  _Float16* xh   = (_Float16*)alloc((size_t)MROWS * DD * 2); // x in f16
  _Float16* hid  = (_Float16*)alloc((size_t)MROWS * DFF * 2);// FFN hidden
  float*    ff   = (float*)   alloc((size_t)MROWS * DD * 4); // FFN2 out

  cvt_f32_f16_kernel<<<(MROWS * DD + 255) / 256, 256, 0, stream>>>(src, hsrc, MROWS * DD);
  auto cvtT = [&](const float* in, _Float16* outT, int K, int N) {
    dim3 g(N / 32, K / 32);
    cvtT_f32_f16_kernel<<<g, 256, 0, stream>>>(in, outT, K, N);
  };
  cvtT(Wq, hWqT, DD, DD);
  cvtT(Wk, hWkT, DD, DD);
  cvtT(Wv, hWvT, DD, DD);
  cvtT(Wo, hWoT, DD, DD);
  cvtT(W1, hW1T, DD, DFF);
  cvtT(W2, hW2T, DFF, DD);

  // ---- Q/K/V projections ----
  dim3 gD(MROWS / 128, DD / 128);
  gemm_f16_kernel<1><<<gD, 256, 0, stream>>>(hsrc, hWqT, bq, nullptr, qb, DD, DD);
  gemm_f16_kernel<1><<<gD, 256, 0, stream>>>(hsrc, hWkT, bk, nullptr, kb_, DD, DD);
  gemm_f16_kernel<1><<<gD, 256, 0, stream>>>(hsrc, hWvT, bv, nullptr, vb, DD, DD);

  // ---- attention (scores output + softmax + P@V) ----
  attn_kernel<<<BB * HH, 192, 0, stream>>>(qb, kb_, vb, matrix, scores_out, aob);

  // ---- output projection ----
  gemm_f16_kernel<0><<<gD, 256, 0, stream>>>(aob, hWoT, bo, proj, nullptr, DD, DD);

  // ---- residual + LN1 (in-place into proj; f16 copy for FFN) ----
  add_ln_kernel<<<MROWS, 256, 0, stream>>>(proj, src, ln1_g, ln1_b, proj, xh);

  // ---- FFN ----
  dim3 gF1(MROWS / 128, DFF / 128);
  gemm_f16_kernel<2><<<gF1, 256, 0, stream>>>(xh, hW1T, b1, nullptr, hid, DFF, DD);
  gemm_f16_kernel<0><<<gD, 256, 0, stream>>>(hid, hW2T, b2, ff, nullptr, DD, DFF);

  // ---- residual + LN2 -> y ----
  add_ln_kernel<<<MROWS, 256, 0, stream>>>(ff, proj, ln2_g, ln2_b, y_out, nullptr);

  (void)in_sizes; (void)n_in; (void)out_size; (void)ws_size;
}
